// UAttention_11321533792267
// MI455X (gfx1250) — compile-verified, matrix operand order checked by reading, not run on hardware
//
#include <hip/hip_runtime.h>

typedef __attribute__((ext_vector_type(16))) _Float16 v16h;
typedef __attribute__((ext_vector_type(8)))  _Float16 v8h;
typedef __attribute__((ext_vector_type(8)))  float    v8f;
typedef int v4i __attribute__((vector_size(16)));
typedef __attribute__((address_space(1))) v4i* gv4i_p;  // global
typedef __attribute__((address_space(3))) v4i* lv4i_p;  // LDS

static constexpr int CB = 4, CS = 1024, CF = 1024, CH = 16, CD = 64;

#if __has_builtin(__builtin_amdgcn_global_load_async_to_lds_b128)
#define ASYNC_LDS 1
#else
#define ASYNC_LDS 0
#endif

static __device__ __forceinline__ void wait_async0() {
#if __has_builtin(__builtin_amdgcn_s_wait_asynccnt)
    __builtin_amdgcn_s_wait_asynccnt(0);
#else
    asm volatile("s_wait_asynccnt 0x0" ::: "memory");
#endif
}

// ---------------------------------------------------------------------------
// Q/K/V projection: Out(f16) = relu(A(f32) @ W(f32) + bias). M=4096, N=K=1024.
// 256 threads = 8 waves; block tile 128x128; K-step 32; double-buffered LDS.
// ---------------------------------------------------------------------------
__global__ __launch_bounds__(256)
void proj_gemm_kernel(const float* __restrict__ A, const float* __restrict__ W,
                      const float* __restrict__ bias, _Float16* __restrict__ Out)
{
    __shared__ __align__(16) _Float16 Ash[2][128][40]; // [buf][m][k], 80B rows
    __shared__ __align__(16) _Float16 Bsh[2][128][40]; // [buf][n][k] (transposed)

    const int tid = threadIdx.x;
    const int m0 = blockIdx.y * 128;
    const int n0 = blockIdx.x * 128;
    const int wave = tid >> 5, lane = tid & 31;
    const int wm = wave >> 2;         // 0..1 -> 64-row slab
    const int wn = wave & 3;          // 0..3 -> 32-col slab
    const int hs = (lane >> 4) << 3;  // 0/8: K-half select (and M+8 in epilogue)
    const int lr = lane & 15;

    v8f acc[4][2] = {};

    // Prologue: stage tile k0=0 into buffer 0.
    #pragma unroll
    for (int i = 0; i < 4; ++i) {
        int seg = tid + i * 256, row = seg >> 3, c4 = (seg & 7) << 2;
        const float4 f = *(const float4*)(A + (size_t)(m0 + row) * CF + c4);
        Ash[0][row][c4 + 0] = (_Float16)f.x;
        Ash[0][row][c4 + 1] = (_Float16)f.y;
        Ash[0][row][c4 + 2] = (_Float16)f.z;
        Ash[0][row][c4 + 3] = (_Float16)f.w;
    }
    #pragma unroll
    for (int i = 0; i < 4; ++i) {
        int seg = tid + i * 256, kr = seg >> 5, n4 = (seg & 31) << 2;
        const float4 f = *(const float4*)(W + (size_t)kr * CF + n0 + n4);
        Bsh[0][n4 + 0][kr] = (_Float16)f.x;
        Bsh[0][n4 + 1][kr] = (_Float16)f.y;
        Bsh[0][n4 + 2][kr] = (_Float16)f.z;
        Bsh[0][n4 + 3][kr] = (_Float16)f.w;
    }
    __syncthreads();

    int nbuf = 0;
    for (int k0 = 0; k0 < CF; k0 += 32) {
        const bool hasnext = (k0 + 32) < CF;
        float4 ra[4], rb[4];
        if (hasnext) {                 // issue next tile's global loads early
            #pragma unroll
            for (int i = 0; i < 4; ++i) {
                int seg = tid + i * 256, row = seg >> 3, c4 = (seg & 7) << 2;
                ra[i] = *(const float4*)(A + (size_t)(m0 + row) * CF + (k0 + 32) + c4);
            }
            #pragma unroll
            for (int i = 0; i < 4; ++i) {
                int seg = tid + i * 256, kr = seg >> 5, n4 = (seg & 31) << 2;
                rb[i] = *(const float4*)(W + (size_t)(k0 + 32 + kr) * CF + n0 + n4);
            }
        }

        v16h a[4], bf[2];
        #pragma unroll
        for (int i = 0; i < 4; ++i) {  // A frag: row M, K = hs..hs+7 | hs+16..+23
            union { v16h v; v8h h[2]; } u;
            const _Float16* p = &Ash[nbuf][wm * 64 + i * 16 + lr][hs];
            u.h[0] = *(const v8h*)p;
            u.h[1] = *(const v8h*)(p + 16);
            a[i] = u.v;
        }
        #pragma unroll
        for (int j = 0; j < 2; ++j) {  // B frag: col N, same K split
            union { v16h v; v8h h[2]; } u;
            const _Float16* p = &Bsh[nbuf][wn * 32 + j * 16 + lr][hs];
            u.h[0] = *(const v8h*)p;
            u.h[1] = *(const v8h*)(p + 16);
            bf[j] = u.v;
        }
        #pragma unroll
        for (int i = 0; i < 4; ++i)
            #pragma unroll
            for (int j = 0; j < 2; ++j)
                acc[i][j] = __builtin_amdgcn_wmma_f32_16x16x32_f16(
                    false, a[i], false, bf[j], (short)0, acc[i][j], false, false);

        if (hasnext) {                 // drain regs into the other buffer
            #pragma unroll
            for (int i = 0; i < 4; ++i) {
                int seg = tid + i * 256, row = seg >> 3, c4 = (seg & 7) << 2;
                Ash[nbuf ^ 1][row][c4 + 0] = (_Float16)ra[i].x;
                Ash[nbuf ^ 1][row][c4 + 1] = (_Float16)ra[i].y;
                Ash[nbuf ^ 1][row][c4 + 2] = (_Float16)ra[i].z;
                Ash[nbuf ^ 1][row][c4 + 3] = (_Float16)ra[i].w;
            }
            #pragma unroll
            for (int i = 0; i < 4; ++i) {
                int seg = tid + i * 256, kr = seg >> 5, n4 = (seg & 31) << 2;
                Bsh[nbuf ^ 1][n4 + 0][kr] = (_Float16)rb[i].x;
                Bsh[nbuf ^ 1][n4 + 1][kr] = (_Float16)rb[i].y;
                Bsh[nbuf ^ 1][n4 + 2][kr] = (_Float16)rb[i].z;
                Bsh[nbuf ^ 1][n4 + 3][kr] = (_Float16)rb[i].w;
            }
        }
        __syncthreads();
        nbuf ^= 1;
    }

    #pragma unroll
    for (int i = 0; i < 4; ++i)
        #pragma unroll
        for (int j = 0; j < 2; ++j) {
            const int n = n0 + wn * 32 + j * 16 + lr;
            const float bv = bias[n];
            #pragma unroll
            for (int r = 0; r < 8; ++r) {
                const int m = m0 + wm * 64 + i * 16 + hs + r;
                float vv = acc[i][j][r] + bv;
                vv = vv > 0.0f ? vv : 0.0f;
                Out[(size_t)m * CF + n] = (_Float16)vv;
            }
        }
}

// ---------------------------------------------------------------------------
// Output projection: out(f32) = x + (ctx(f16) @ Wo + bo) * gamma.
// A operand is already f16; staged via async-to-LDS (ASYNCcnt) when available.
// ---------------------------------------------------------------------------
__global__ __launch_bounds__(256)
void out_gemm_kernel(const _Float16* __restrict__ A, const float* __restrict__ W,
                     const float* __restrict__ bias, const float* __restrict__ X,
                     const float* __restrict__ gamma, float* __restrict__ Out)
{
    __shared__ __align__(16) _Float16 Ash[2][128][40];
    __shared__ __align__(16) _Float16 Bsh[2][128][40];

    const int tid = threadIdx.x;
    const int m0 = blockIdx.y * 128;
    const int n0 = blockIdx.x * 128;
    const int wave = tid >> 5, lane = tid & 31;
    const int wm = wave >> 2, wn = wave & 3;
    const int hs = (lane >> 4) << 3;
    const int lr = lane & 15;

    v8f acc[4][2] = {};

    #pragma unroll
    for (int i = 0; i < 2; ++i) {      // prologue A (f16): 128x32
        int seg = tid + i * 256, row = seg >> 2, c8 = (seg & 3) << 3;
        v8h hv = *(const v8h*)(A + (size_t)(m0 + row) * CF + c8);
        *(v8h*)&Ash[0][row][c8] = hv;
    }
    #pragma unroll
    for (int i = 0; i < 4; ++i) {      // prologue B transposed
        int seg = tid + i * 256, kr = seg >> 5, n4 = (seg & 31) << 2;
        const float4 f = *(const float4*)(W + (size_t)kr * CF + n0 + n4);
        Bsh[0][n4 + 0][kr] = (_Float16)f.x;
        Bsh[0][n4 + 1][kr] = (_Float16)f.y;
        Bsh[0][n4 + 2][kr] = (_Float16)f.z;
        Bsh[0][n4 + 3][kr] = (_Float16)f.w;
    }
    __syncthreads();

    int nbuf = 0;
    for (int k0 = 0; k0 < CF; k0 += 32) {
        const bool hasnext = (k0 + 32) < CF;
        float4 rb[4];
#if ASYNC_LDS
        if (hasnext) {                 // ASYNCcnt-tracked direct global->LDS
            #pragma unroll
            for (int i = 0; i < 2; ++i) {
                int seg = tid + i * 256, row = seg >> 2, c8 = (seg & 3) << 3;
                __builtin_amdgcn_global_load_async_to_lds_b128(
                    (gv4i_p)(A + (size_t)(m0 + row) * CF + (k0 + 32) + c8),
                    (lv4i_p)&Ash[nbuf ^ 1][row][c8],
                    0, 0);
            }
        }
#else
        v8h rah[2];
        if (hasnext) {
            #pragma unroll
            for (int i = 0; i < 2; ++i) {
                int seg = tid + i * 256, row = seg >> 2, c8 = (seg & 3) << 3;
                rah[i] = *(const v8h*)(A + (size_t)(m0 + row) * CF + (k0 + 32) + c8);
            }
        }
#endif
        if (hasnext) {
            #pragma unroll
            for (int i = 0; i < 4; ++i) {
                int seg = tid + i * 256, kr = seg >> 5, n4 = (seg & 31) << 2;
                rb[i] = *(const float4*)(W + (size_t)(k0 + 32 + kr) * CF + n0 + n4);
            }
        }

        v16h a[4], bf[2];
        #pragma unroll
        for (int i = 0; i < 4; ++i) {
            union { v16h v; v8h h[2]; } u;
            const _Float16* p = &Ash[nbuf][wm * 64 + i * 16 + lr][hs];
            u.h[0] = *(const v8h*)p;
            u.h[1] = *(const v8h*)(p + 16);
            a[i] = u.v;
        }
        #pragma unroll
        for (int j = 0; j < 2; ++j) {
            union { v16h v; v8h h[2]; } u;
            const _Float16* p = &Bsh[nbuf][wn * 32 + j * 16 + lr][hs];
            u.h[0] = *(const v8h*)p;
            u.h[1] = *(const v8h*)(p + 16);
            bf[j] = u.v;
        }
        #pragma unroll
        for (int i = 0; i < 4; ++i)
            #pragma unroll
            for (int j = 0; j < 2; ++j)
                acc[i][j] = __builtin_amdgcn_wmma_f32_16x16x32_f16(
                    false, a[i], false, bf[j], (short)0, acc[i][j], false, false);

        if (hasnext) {
#if !ASYNC_LDS
            #pragma unroll
            for (int i = 0; i < 2; ++i) {
                int seg = tid + i * 256, row = seg >> 2, c8 = (seg & 3) << 3;
                *(v8h*)&Ash[nbuf ^ 1][row][c8] = rah[i];
            }
#endif
            #pragma unroll
            for (int i = 0; i < 4; ++i) {
                int seg = tid + i * 256, kr = seg >> 5, n4 = (seg & 31) << 2;
                Bsh[nbuf ^ 1][n4 + 0][kr] = (_Float16)rb[i].x;
                Bsh[nbuf ^ 1][n4 + 1][kr] = (_Float16)rb[i].y;
                Bsh[nbuf ^ 1][n4 + 2][kr] = (_Float16)rb[i].z;
                Bsh[nbuf ^ 1][n4 + 3][kr] = (_Float16)rb[i].w;
            }
        }
#if ASYNC_LDS
        wait_async0();
#endif
        __syncthreads();
        nbuf ^= 1;
    }

    const float g = gamma[0];
    #pragma unroll
    for (int i = 0; i < 4; ++i)
        #pragma unroll
        for (int j = 0; j < 2; ++j) {
            const int n = n0 + wn * 32 + j * 16 + lr;
            const float bv = bias[n];
            #pragma unroll
            for (int r = 0; r < 8; ++r) {
                const int m = m0 + wm * 64 + i * 16 + hs + r;
                Out[(size_t)m * CF + n] = X[(size_t)m * CF + n] + (acc[i][j][r] + bv) * g;
            }
        }
}

// ---------------------------------------------------------------------------
// logits[bh,m,n] = (Q[b,m,h,:] . K[b,n,h,:]) / 8, written into d_out's attn
// region. Row-major f16 Q/K rows are exactly the WMMA lane layout, so
// fragments load straight from global (Q/K/V total 24 MB -> lives in 192MB L2).
// ---------------------------------------------------------------------------
__global__ __launch_bounds__(256)
void logits_kernel(const _Float16* __restrict__ Q, const _Float16* __restrict__ K,
                   float* __restrict__ attn)
{
    const int tid = threadIdx.x;
    const int bh = blockIdx.z;
    const int b = bh >> 4, h = bh & 15;
    const int m0 = blockIdx.y * 128, n0 = blockIdx.x * 128;
    const int wave = tid >> 5, lane = tid & 31;
    const int wm = wave >> 2, wn = wave & 3;
    const int hs = (lane >> 4) << 3;
    const int lr = lane & 15;

    const _Float16* qb = Q + (size_t)b * CS * CF + h * CD;
    const _Float16* kb = K + (size_t)b * CS * CF + h * CD;

    v8f acc[4][2] = {};
    #pragma unroll
    for (int k0 = 0; k0 < CD; k0 += 32) {
        v16h a[4], bf[2];
        #pragma unroll
        for (int i = 0; i < 4; ++i) {
            union { v16h v; v8h h[2]; } u;
            const _Float16* p = qb + (size_t)(m0 + wm * 64 + i * 16 + lr) * CF + k0 + hs;
            u.h[0] = *(const v8h*)p;
            u.h[1] = *(const v8h*)(p + 16);
            a[i] = u.v;
        }
        #pragma unroll
        for (int j = 0; j < 2; ++j) {
            union { v16h v; v8h h[2]; } u;
            const _Float16* p = kb + (size_t)(n0 + wn * 32 + j * 16 + lr) * CF + k0 + hs;
            u.h[0] = *(const v8h*)p;
            u.h[1] = *(const v8h*)(p + 16);
            bf[j] = u.v;
        }
        #pragma unroll
        for (int i = 0; i < 4; ++i)
            #pragma unroll
            for (int j = 0; j < 2; ++j)
                acc[i][j] = __builtin_amdgcn_wmma_f32_16x16x32_f16(
                    false, a[i], false, bf[j], (short)0, acc[i][j], false, false);
    }

    const float scl = 0.125f; // 1/sqrt(64)
    #pragma unroll
    for (int i = 0; i < 4; ++i)
        #pragma unroll
        for (int j = 0; j < 2; ++j) {
            const int n = n0 + wn * 32 + j * 16 + lr;
            #pragma unroll
            for (int r = 0; r < 8; ++r) {
                const int m = m0 + wm * 64 + i * 16 + hs + r;
                attn[((size_t)bh * CS + m) * CS + n] = acc[i][j][r] * scl;
            }
        }
}

// ---------------------------------------------------------------------------
// In-place row softmax (1024 cols). One 256-thread block per row; b128 access.
// This kernel is the HBM-traffic peak (512 MB), so keep it pure streaming.
// ---------------------------------------------------------------------------
__global__ __launch_bounds__(256)
void softmax_kernel(float* __restrict__ attn)
{
    __shared__ float sb[256];
    const int tid = threadIdx.x;
    float* row = attn + (size_t)blockIdx.x * CS;

    float4 f = *(const float4*)(row + tid * 4);
    float mx = fmaxf(fmaxf(f.x, f.y), fmaxf(f.z, f.w));
    sb[tid] = mx; __syncthreads();
    for (int s = 128; s > 0; s >>= 1) {
        if (tid < s) sb[tid] = fmaxf(sb[tid], sb[tid + s]);
        __syncthreads();
    }
    const float rmax = sb[0];
    __syncthreads();

    f.x = __expf(f.x - rmax);
    f.y = __expf(f.y - rmax);
    f.z = __expf(f.z - rmax);
    f.w = __expf(f.w - rmax);
    sb[tid] = f.x + f.y + f.z + f.w; __syncthreads();
    for (int s = 128; s > 0; s >>= 1) {
        if (tid < s) sb[tid] += sb[tid + s];
        __syncthreads();
    }
    const float inv = 1.0f / sb[0];

    f.x *= inv; f.y *= inv; f.z *= inv; f.w *= inv;
    *(float4*)(row + tid * 4) = f;
}

// ---------------------------------------------------------------------------
// ctx[b,m,h,:] = attn(f32, d_out) @ V(f16). Tile 128(M) x 64(N=D), K-step 32,
// double-buffered V tile in LDS, attn converted f32->f16 in-register,
// prefetch on the streamed attn rows.
// ---------------------------------------------------------------------------
__global__ __launch_bounds__(256)
void ctx_kernel(const float* __restrict__ attn, const _Float16* __restrict__ V,
                _Float16* __restrict__ ctx)
{
    __shared__ __align__(16) _Float16 Vsh[2][64][40]; // [buf][d][k]
    const int tid = threadIdx.x;
    const int bh = blockIdx.z;
    const int b = bh >> 4, h = bh & 15;
    const int m0 = blockIdx.y * 128;
    const int wave = tid >> 5, lane = tid & 31;
    const int wm = wave >> 1;  // 0..3 -> 32-row slab
    const int wn = wave & 1;   // 0..1 -> 32-col slab
    const int hs = (lane >> 4) << 3;
    const int lr = lane & 15;

    const _Float16* vb = V + (size_t)b * CS * CF + h * CD;
    const float* ab = attn + (size_t)bh * CS * CS;

    const int kr = tid >> 3;        // 0..31
    const int n8 = (tid & 7) << 3;  // 0..56

    {   // prologue: V tile k0=0 -> buffer 0 (transposed)
        v8h hv = *(const v8h*)(vb + (size_t)kr * CF + n8);
        #pragma unroll
        for (int j = 0; j < 8; ++j) Vsh[0][n8 + j][kr] = hv[j];
    }
    __syncthreads();

    v8f acc[2][2] = {};
    int nbuf = 0;
    for (int k0 = 0; k0 < CS; k0 += 32) {
        const bool hasnext = (k0 + 32) < CS;
        v8h hvn;
        if (hasnext) hvn = *(const v8h*)(vb + (size_t)(k0 + 32 + kr) * CF + n8);
        if (k0 + 64 < CS)
            __builtin_prefetch(ab + (size_t)(m0 + wm * 32 + lr) * CS + k0 + 64, 0, 3);

        v16h a[2], bf[2];
        #pragma unroll
        for (int i = 0; i < 2; ++i) {   // attn fragment: f32 -> f16 in-register
            union { v16h v; _Float16 e[16]; } u;
            const float* p = ab + (size_t)(m0 + wm * 32 + i * 16 + lr) * CS + k0 + hs;
            #pragma unroll
            for (int t = 0; t < 8; ++t) u.e[t] = (_Float16)p[t];
            #pragma unroll
            for (int t = 0; t < 8; ++t) u.e[8 + t] = (_Float16)p[16 + t];
            a[i] = u.v;
        }
        #pragma unroll
        for (int j = 0; j < 2; ++j) {
            union { v16h v; v8h h[2]; } u;
            const _Float16* p = &Vsh[nbuf][wn * 32 + j * 16 + lr][hs];
            u.h[0] = *(const v8h*)p;
            u.h[1] = *(const v8h*)(p + 16);
            bf[j] = u.v;
        }
        #pragma unroll
        for (int i = 0; i < 2; ++i)
            #pragma unroll
            for (int j = 0; j < 2; ++j)
                acc[i][j] = __builtin_amdgcn_wmma_f32_16x16x32_f16(
                    false, a[i], false, bf[j], (short)0, acc[i][j], false, false);

        if (hasnext) {
            #pragma unroll
            for (int j = 0; j < 8; ++j) Vsh[nbuf ^ 1][n8 + j][kr] = hvn[j];
        }
        __syncthreads();
        nbuf ^= 1;
    }

    #pragma unroll
    for (int i = 0; i < 2; ++i)
        #pragma unroll
        for (int j = 0; j < 2; ++j) {
            const int n = wn * 32 + j * 16 + lr; // d index 0..63
            #pragma unroll
            for (int r = 0; r < 8; ++r) {
                const int m = m0 + wm * 32 + i * 16 + hs + r;
                ctx[((size_t)b * CS + m) * CF + h * CD + n] = (_Float16)acc[i][j][r];
            }
        }
}

extern "C" void kernel_launch(void* const* d_in, const int* in_sizes, int n_in,
                              void* d_out, int out_size, void* d_ws, size_t ws_size,
                              hipStream_t stream) {
    (void)in_sizes; (void)n_in; (void)out_size; (void)ws_size;
    const float* x   = (const float*)d_in[0];
    const float* kin = (const float*)d_in[1];
    const float* Wq  = (const float*)d_in[2];  const float* bq = (const float*)d_in[3];
    const float* Wk  = (const float*)d_in[4];  const float* bk = (const float*)d_in[5];
    const float* Wv  = (const float*)d_in[6];  const float* bv = (const float*)d_in[7];
    const float* Wo  = (const float*)d_in[8];  const float* bo = (const float*)d_in[9];
    const float* gamma = (const float*)d_in[10];

    float* out  = (float*)d_out;
    float* attn = out + (size_t)CB * CS * CF;   // tuple: out (4M f32) then attn (64M f32)

    _Float16* qf = (_Float16*)d_ws;             // 8 MB each
    _Float16* kf = qf + (size_t)CB * CS * CF;
    _Float16* vf = kf + (size_t)CB * CS * CF;
    _Float16* cf = vf + (size_t)CB * CS * CF;

    dim3 blk(256);
    dim3 gProj(CF / 128, (CB * CS) / 128);      // (8, 32)
    proj_gemm_kernel<<<gProj, blk, 0, stream>>>(x,   Wq, bq, qf);
    proj_gemm_kernel<<<gProj, blk, 0, stream>>>(kin, Wk, bk, kf);
    proj_gemm_kernel<<<gProj, blk, 0, stream>>>(kin, Wv, bv, vf);

    dim3 gLog(CS / 128, CS / 128, CB * CH);     // (8, 8, 64)
    logits_kernel<<<gLog, blk, 0, stream>>>(qf, kf, attn);

    dim3 gSm(CB * CH * CS);                     // 65536 rows
    softmax_kernel<<<gSm, blk, 0, stream>>>(attn);

    dim3 gCtx(1, CS / 128, CB * CH);            // (1, 8, 64)
    ctx_kernel<<<gCtx, blk, 0, stream>>>(attn, vf, cf);

    out_gemm_kernel<<<gProj, blk, 0, stream>>>(cf, Wo, bo, x, gamma, out);
}
